// Attention_84456236909284
// MI455X (gfx1250) — compile-verified, matrix operand order checked by reading, not run on hardware
//
#include <hip/hip_runtime.h>
#include <hip/hip_bf16.h>

// ---------------------------------------------------------------------------
// Attention (Bahdanau-style) for MI455X / gfx1250, wave32.
// Big GEMM (M=50176, K=2048, N=512) on v_wmma_f32_16x16x32_bf16, f32 accum.
//   * 4 batches per workgroup: each B-fragment register set feeds 16 WMMAs,
//     cutting W_enc L2 traffic 4x (6.6 GB -> 1.65 GB) -- L2, not HBM, is the
//     binding roof for this kernel.
//   * dec_comb / W_full epilogue operands staged via
//     GLOBAL_LOAD_ASYNC_TO_LDS_B128 at block start (ASYNCcnt path), waited
//     with s_wait_asynccnt only after the K loop: full overlap.
// ---------------------------------------------------------------------------

typedef __attribute__((ext_vector_type(16))) __bf16 v16bf;
typedef __attribute__((ext_vector_type(8)))  __bf16 v8bf;
typedef __attribute__((ext_vector_type(8)))  float  v8f;
typedef __attribute__((ext_vector_type(4)))  int    v4i;

#define BATCH   256
#define ENC_C   2048
#define DEC_DIM 512
#define ATT_DIM 512
#define PIX     196
#define PPAD    208            // 13 tiles of 16 rows
#define MTILES  13
#define NB      4              // batches per workgroup in the GEMM kernel

// --------------------------- async-copy helpers ----------------------------
// ROCm 7.2 signature (from hipcc diagnostic): param0 = v4i addrspace(1)*
// (printed as "int __vector(4) __device__ *"), param1 = v4i addrspace(3)*,
// then two immediate ints (offset, cpol).
__device__ __forceinline__ void async_copy_b128(const void* gsrc, unsigned lds_off) {
#if __has_builtin(__builtin_amdgcn_global_load_async_to_lds_b128)
    typedef __attribute__((address_space(1))) v4i* gptr_t;
    typedef __attribute__((address_space(3))) v4i* lptr_t;
    __builtin_amdgcn_global_load_async_to_lds_b128(
        (gptr_t)(unsigned long long)(size_t)gsrc,
        (lptr_t)(size_t)lds_off, 0, 0);
#else
    asm volatile("global_load_async_to_lds_b128 %0, %1, off"
                 :: "v"(lds_off), "v"(gsrc) : "memory");
#endif
}

__device__ __forceinline__ void wait_async_zero() {
#if __has_builtin(__builtin_amdgcn_s_wait_asynccnt)
    __builtin_amdgcn_s_wait_asynccnt(0);
#else
    asm volatile("s_wait_asynccnt 0x0" ::: "memory");
#endif
}

// ---------------------------------------------------------------------------
// Kernel 1: W_enc [K=2048][A=512] f32  ->  Wt [A=512][K=2048] bf16
// ---------------------------------------------------------------------------
__global__ __launch_bounds__(256)
void wenc_transpose_bf16(const float* __restrict__ We, __bf16* __restrict__ wt) {
    int idx = blockIdx.x * 256 + threadIdx.x;   // 0 .. 2048*512-1
    int a = idx >> 11;                          // / ENC_C
    int k = idx & (ENC_C - 1);
    wt[idx] = (__bf16)We[(size_t)k * ATT_DIM + a];
}

// ---------------------------------------------------------------------------
// Kernel 2: dec_comb[b][a] = dh[b] @ W_dec[:,a] + b_dec[a] + b_enc[a]
// ---------------------------------------------------------------------------
__global__ __launch_bounds__(512)
void dec_proj_kernel(const float* __restrict__ dh, const float* __restrict__ Wd,
                     const float* __restrict__ bdec, const float* __restrict__ benc,
                     float* __restrict__ out) {
    int b = blockIdx.x;
    int a = threadIdx.x;                         // 512 threads = 16 waves
    float acc = bdec[a] + benc[a];
    const float* dhr = dh + (size_t)b * DEC_DIM;
    for (int k = 0; k < DEC_DIM; ++k)
        acc = fmaf(dhr[k], Wd[(size_t)k * ATT_DIM + a], acc);
    out[(size_t)b * ATT_DIM + a] = acc;
}

// ---------------------------------------------------------------------------
// Kernel 3: WMMA logits. Workgroup = (16-pixel M-tile, 4 consecutive batches).
// 8 waves; wave w owns N columns [w*64, w*64+64) for all 4 batches:
// per K-chunk(32): 4 B-fragment sets loaded once, 4 A-fragments built,
// 16 v_wmma_f32_16x16x32_bf16. Epilogue fuses +dec_comb, relu, *W_full.
// b_full omitted: constant logit shift cancels in softmax.
// ---------------------------------------------------------------------------
__global__ __launch_bounds__(256)
void att_logits_kernel(const float*  __restrict__ enc,       // [B][C][196]
                       const __bf16* __restrict__ wt,        // [A=512][K=2048]
                       const float*  __restrict__ dec_comb,  // [B][512]
                       const float*  __restrict__ wfull,     // [512]
                       float*        __restrict__ att)       // [B][PPAD]
{
    const int mtile = blockIdx.x;          // 0..12
    const int b0    = blockIdx.y * NB;     // first batch of this group
    const int tid   = threadIdx.x;
    const int lane  = tid & 31;
    const int wave  = tid >> 5;

    __shared__ float smem_dec[NB][ATT_DIM];    // 8 KB
    __shared__ float smem_wf[ATT_DIM];         // 2 KB
    __shared__ float lds_part[NB][8][16];      // 2 KB

    // ---- stage epilogue operands through the async-to-LDS path; these are
    //      not needed until after the K loop, so the DMA fully overlaps. ----
    {
        const unsigned dec_off = (unsigned)(size_t)&smem_dec[0][0];
        const unsigned wf_off  = (unsigned)(size_t)&smem_wf[0];
        const float*   dsrc    = dec_comb + (size_t)b0 * ATT_DIM;
        for (int idx = tid; idx < (NB * ATT_DIM + ATT_DIM) / 4; idx += 256) {
            if (idx < (NB * ATT_DIM) / 4)
                async_copy_b128(dsrc + idx * 4, dec_off + idx * 16);
            else {
                int widx = idx - (NB * ATT_DIM) / 4;
                async_copy_b128(wfull + widx * 4, wf_off + widx * 16);
            }
        }
    }

    // --- A-fragment lane mapping (16-bit A 16x32): lane l -> m = l&15,
    //     ko = (l>>4)*8; elements [0..7] = K ko..ko+7, [8..15] = K 16+ko.. ---
    const int m16 = lane & 15;
    const int ko  = (lane >> 4) * 8;
    int p = mtile * 16 + m16;
    if (p > PIX - 1) p = PIX - 1;          // clamp tail rows; masked in softmax

    const float* ap[NB];
#pragma unroll
    for (int bt = 0; bt < NB; ++bt)
        ap[bt] = enc + (size_t)(b0 + bt) * ENC_C * PIX + p;  // elem k at ap[k*196]

    const int n16   = lane & 15;
    const int ncol0 = wave * 64;           // this wave's first N column

    v8f acc[NB][4] = {};

    for (int kb = 0; kb < ENC_C; kb += 32) {
        // B fragments: one load per n-tile, reused by all NB batches.
        v16bf bfrag[4];
#pragma unroll
        for (int t = 0; t < 4; ++t) {
            const __bf16* bp = wt + (size_t)(ncol0 + t * 16 + n16) * ENC_C + (kb + ko);
            v8bf lo = *(const v8bf*)bp;
            v8bf hi = *(const v8bf*)(bp + 16);
#pragma unroll
            for (int i = 0; i < 8; ++i) { bfrag[t][i] = lo[i]; bfrag[t][i + 8] = hi[i]; }
        }

#pragma unroll
        for (int bt = 0; bt < NB; ++bt) {
            // A fragment: 16 strided dword loads (64B-coalesced per half-wave),
            // converted f32 -> bf16 in registers.
            v16bf afrag;
#pragma unroll
            for (int i = 0; i < 8; ++i) {
                afrag[i]     = (__bf16)ap[bt][(size_t)(kb + ko + i) * PIX];
                afrag[i + 8] = (__bf16)ap[bt][(size_t)(kb + 16 + ko + i) * PIX];
            }
            __builtin_prefetch(ap[bt] + (size_t)(kb + 32 + ko) * PIX, 0, 1);

#pragma unroll
            for (int t = 0; t < 4; ++t)
                acc[bt][t] = __builtin_amdgcn_wmma_f32_16x16x32_bf16(
                    false, afrag, false, bfrag[t], (short)0, acc[bt][t], false, false);
        }
    }

    // epilogue operands must be in LDS by now
    wait_async_zero();
    __syncthreads();

    // --- logit_m += relu(acc[m][n] + dec_comb[b][n]) * wfull[n] ---
    // C/D layout: vgpr r, lane l -> m = (l>>4)*8 + r, n = l&15.
#pragma unroll
    for (int bt = 0; bt < NB; ++bt) {
        float part[8];
#pragma unroll
        for (int r = 0; r < 8; ++r) part[r] = 0.0f;
#pragma unroll
        for (int t = 0; t < 4; ++t) {
            const int ncol = ncol0 + t * 16 + n16;
            const float dterm = smem_dec[bt][ncol];
            const float wfv   = smem_wf[ncol];
#pragma unroll
            for (int r = 0; r < 8; ++r) {
                float v = acc[bt][t][r] + dterm;
                v = v > 0.0f ? v : 0.0f;
                part[r] = fmaf(v, wfv, part[r]);
            }
        }
        // deterministic reduce over the 16 lanes sharing each m
#pragma unroll
        for (int r = 0; r < 8; ++r) {
            float v = part[r];
            v += __shfl_xor(v, 1, 32);
            v += __shfl_xor(v, 2, 32);
            v += __shfl_xor(v, 4, 32);
            v += __shfl_xor(v, 8, 32);
            if ((lane & 15) == 0) lds_part[bt][wave][(lane >> 4) * 8 + r] = v;
        }
    }
    __syncthreads();

    if (tid < NB * 16) {
        const int bt = tid >> 4;
        const int m  = tid & 15;
        float s = 0.0f;
#pragma unroll
        for (int w = 0; w < 8; ++w) s += lds_part[bt][w][m];   // fixed order
        att[(size_t)(b0 + bt) * PPAD + mtile * 16 + m] = s;
    }
}

// ---------------------------------------------------------------------------
// Kernel 4: softmax over 196 pixels per batch (pads masked out)
// ---------------------------------------------------------------------------
__global__ __launch_bounds__(256)
void softmax_kernel(const float* __restrict__ att, float* __restrict__ alpha) {
    const int b = blockIdx.x;
    const int t = threadIdx.x;
    __shared__ float red[256];

    float x = (t < PIX) ? att[(size_t)b * PPAD + t] : -3.0e38f;
    red[t] = x;
    __syncthreads();
    for (int s = 128; s > 0; s >>= 1) {
        if (t < s) red[t] = fmaxf(red[t], red[t + s]);
        __syncthreads();
    }
    const float mx = red[0];
    __syncthreads();

    float e = (t < PIX) ? __expf(x - mx) : 0.0f;
    red[t] = e;
    __syncthreads();
    for (int s = 128; s > 0; s >>= 1) {
        if (t < s) red[t] += red[t + s];
        __syncthreads();
    }
    if (t < PIX) alpha[(size_t)b * PIX + t] = e * (1.0f / red[0]);
}

// ---------------------------------------------------------------------------
// Kernel 5: context[b][c] = sum_p enc[b][c][p] * alpha[b][p]
// One wave per (b,c): 196 contiguous floats, fully coalesced; pure-BW pass.
// ---------------------------------------------------------------------------
__global__ __launch_bounds__(256)
void context_kernel(const float* __restrict__ enc, const float* __restrict__ alpha,
                    float* __restrict__ ctx) {
    const int w    = (int)((blockIdx.x * blockDim.x + threadIdx.x) >> 5);
    const int lane = threadIdx.x & 31;
    const int b    = w >> 11;              // 2048 channels per batch
    const int c    = w & (ENC_C - 1);

    const float* row = enc + ((size_t)b * ENC_C + c) * PIX;
    const float* al  = alpha + (size_t)b * PIX;
    float s = 0.0f;
    for (int pp = lane; pp < PIX; pp += 32) s = fmaf(row[pp], al[pp], s);

    s += __shfl_xor(s, 16, 32);
    s += __shfl_xor(s, 8, 32);
    s += __shfl_xor(s, 4, 32);
    s += __shfl_xor(s, 2, 32);
    s += __shfl_xor(s, 1, 32);
    if (lane == 0) ctx[(size_t)b * ENC_C + c] = s;
}

// ---------------------------------------------------------------------------
extern "C" void kernel_launch(void* const* d_in, const int* in_sizes, int n_in,
                              void* d_out, int out_size, void* d_ws, size_t ws_size,
                              hipStream_t stream) {
    const float* enc    = (const float*)d_in[0];   // [256,2048,14,14]
    const float* dh     = (const float*)d_in[1];   // [256,512]
    const float* W_enc  = (const float*)d_in[2];   // [2048,512]
    const float* b_enc  = (const float*)d_in[3];   // [512]
    const float* W_dec  = (const float*)d_in[4];   // [512,512]
    const float* b_dec  = (const float*)d_in[5];   // [512]
    const float* W_full = (const float*)d_in[6];   // [512,1]
    // d_in[7] = b_full: constant logit shift, cancels in softmax.

    float* ctx_out   = (float*)d_out;                          // [256,2048]
    float* alpha_out = (float*)d_out + (size_t)BATCH * ENC_C;  // [256,196]

    // workspace: Wt bf16 (2 MB) | dec_comb f32 (512 KB) | att f32 (208 KB)
    __bf16* wt      = (__bf16*)d_ws;
    float* dec_comb = (float*)((char*)d_ws + (size_t)ATT_DIM * ENC_C * sizeof(__bf16));
    float* att      = dec_comb + (size_t)BATCH * ATT_DIM;

    wenc_transpose_bf16<<<(ENC_C * ATT_DIM) / 256, 256, 0, stream>>>(W_enc, wt);
    dec_proj_kernel<<<BATCH, 512, 0, stream>>>(dh, W_dec, b_dec, b_enc, dec_comb);
    att_logits_kernel<<<dim3(MTILES, BATCH / NB), 256, 0, stream>>>(enc, wt, dec_comb,
                                                                    W_full, att);
    softmax_kernel<<<BATCH, 256, 0, stream>>>(att, alpha_out);
    context_kernel<<<(BATCH * ENC_C) / 8, 256, 0, stream>>>(enc, alpha_out, ctx_out);
}
